// gcn_32762010534300
// MI455X (gfx1250) — matrix operand order repaired and checked by reading, not compiled, not run on hardware
//
#include <hip/hip_runtime.h>
#include <hip/hip_bf16.h>

// MI455X / gfx1250: wave32, WMMA 16x16x32 bf16 with f32 accumulation.
// GCN: out = softmax(A @ relu(A @ (X@W1)) @ W2), A streamed f32->bf16 in-register.

typedef __attribute__((ext_vector_type(16))) __bf16 v16bf;
typedef __attribute__((ext_vector_type(8)))  __bf16 v8bf;
typedef __attribute__((ext_vector_type(8)))  float  v8f;

constexpr int kN   = 20000;
constexpr int kIPI = 128;
constexpr int kOPI = 64;
constexpr int kOPO = 16;
constexpr int kRowTiles = kN / 16;          // 1250
constexpr int kWavesPerBlock = 8;           // 256 threads

// ---- A-fragment loader: 16x32 bf16 A operand from row-major f32 matrix ----
// Layout: lanes 0-15 (row = lane) hold K {0..15} in elems 0-15; lanes 16-31
// (row = lane-16) hold K {16..31}.
// Caller passes ap = &M[row*ld + k0 + (lane>>4)*16]; we read 16 consecutive floats.
__device__ __forceinline__ v16bf load_a_frag_f32(const float* __restrict__ ap) {
  float4 x0 = ((const float4*)ap)[0];
  float4 x1 = ((const float4*)ap)[1];
  float4 y0 = ((const float4*)(ap + 8))[0];
  float4 y1 = ((const float4*)(ap + 8))[1];
  v16bf a;
  a[0]  = (__bf16)x0.x; a[1]  = (__bf16)x0.y; a[2]  = (__bf16)x0.z; a[3]  = (__bf16)x0.w;
  a[4]  = (__bf16)x1.x; a[5]  = (__bf16)x1.y; a[6]  = (__bf16)x1.z; a[7]  = (__bf16)x1.w;
  a[8]  = (__bf16)y0.x; a[9]  = (__bf16)y0.y; a[10] = (__bf16)y0.z; a[11] = (__bf16)y0.w;
  a[12] = (__bf16)y1.x; a[13] = (__bf16)y1.y; a[14] = (__bf16)y1.z; a[15] = (__bf16)y1.w;
  return a;
}

// ---------------- Stage 1: XW1^T[o][i] = (features @ W1)[i][o] in bf16 ------
__global__ __launch_bounds__(256) void gcn_xw1_kernel(
    const float* __restrict__ feat, const float* __restrict__ W1,
    __bf16* __restrict__ xw1t) {
  __shared__ float tile[16 * kIPI];                 // 8 KB feature tile
  const int i0 = blockIdx.x * 16;
  const int t  = threadIdx.x;
  for (int idx = t; idx < 16 * kIPI; idx += 256) {
    int r = idx >> 7, c = idx & (kIPI - 1);
    tile[idx] = feat[(size_t)(i0 + r) * kIPI + c];
  }
  __syncthreads();
  const int row   = t >> 4;                         // 0..15
  const int cbase = t & 15;                         // 0..15 (cols cbase+16j)
  float acc[4] = {0.f, 0.f, 0.f, 0.f};
  for (int k = 0; k < kIPI; ++k) {
    float f = tile[row * kIPI + k];
    const float* w = W1 + (size_t)k * kOPI + cbase;
    acc[0] = fmaf(f, w[0],  acc[0]);
    acc[1] = fmaf(f, w[16], acc[1]);
    acc[2] = fmaf(f, w[32], acc[2]);
    acc[3] = fmaf(f, w[48], acc[3]);
  }
  for (int j = 0; j < 4; ++j) {
    int o = cbase + 16 * j;
    xw1t[(size_t)o * kN + i0 + row] = (__bf16)acc[j];
  }
}

// -- Stage 2: per-wave 16-row tile: h = relu(A @ XW1); G^T = (h @ W2)^T bf16 --
__global__ __launch_bounds__(256) void gcn_layer1_kernel(
    const float* __restrict__ A, const __bf16* __restrict__ xw1t,
    const float* __restrict__ W2, __bf16* __restrict__ gt) {
  __shared__ __bf16 hbuf[kWavesPerBlock][16 * kOPI];   // 2 KB per wave
  const int wave = threadIdx.x >> 5;
  const int lane = threadIdx.x & 31;
  const int tileIdx = blockIdx.x * kWavesPerBlock + wave;
  if (tileIdx >= kRowTiles) return;                    // wave-uniform exit
  const int m0 = tileIdx * 16;
  const int lo = lane & 15;
  const int hi = lane >> 4;

  // Loop-invariant W2 B-fragments (64x16 split into two 32x16 K-chunks).
  // B layout: lanes 0-15 hold K k0..k0+15 (elem t = K k0+t), lanes 16-31 K+16.
  v16bf w2f[2];
  for (int c = 0; c < 2; ++c) {
    int kb = c * 32 + hi * 16;
    for (int t = 0; t < 16; ++t)
      w2f[c][t] = (__bf16)W2[(size_t)(kb + t) * kOPO + lo];
  }

  v8f acc0 = {}, acc1 = {}, acc2 = {}, acc3 = {};
  const float* arow = A + (size_t)(m0 + lo) * kN + hi * 16;
  const __bf16* brow = xw1t + (size_t)lo * kN + hi * 16;
  for (int k0 = 0; k0 < kN; k0 += 32) {
    v16bf af = load_a_frag_f32(arow + k0);
    v16bf b0 = *(const v16bf*)(brow + k0);
    v16bf b1 = *(const v16bf*)(brow + k0 + (size_t)16 * kN);
    v16bf b2 = *(const v16bf*)(brow + k0 + (size_t)32 * kN);
    v16bf b3 = *(const v16bf*)(brow + k0 + (size_t)48 * kN);
    acc0 = __builtin_amdgcn_wmma_f32_16x16x32_bf16(false, af, false, b0, (short)0, acc0, false, false);
    acc1 = __builtin_amdgcn_wmma_f32_16x16x32_bf16(false, af, false, b1, (short)0, acc1, false, false);
    acc2 = __builtin_amdgcn_wmma_f32_16x16x32_bf16(false, af, false, b2, (short)0, acc2, false, false);
    acc3 = __builtin_amdgcn_wmma_f32_16x16x32_bf16(false, af, false, b3, (short)0, acc3, false, false);
  }

  // relu + stash h tile (16x64, row-major) into this wave's private LDS slice.
  // C-frag layout: elem r <-> row (r + 8*hi), col lo (+16j per n-tile).
  __bf16* h = &hbuf[wave][0];
  for (int r = 0; r < 8; ++r) {
    int m = r + 8 * hi;
    h[m * kOPI + lo +  0] = (__bf16)fmaxf(acc0[r], 0.f);
    h[m * kOPI + lo + 16] = (__bf16)fmaxf(acc1[r], 0.f);
    h[m * kOPI + lo + 32] = (__bf16)fmaxf(acc2[r], 0.f);
    h[m * kOPI + lo + 48] = (__bf16)fmaxf(acc3[r], 0.f);
  }
  // Same-wave LDS RAW: compiler inserts s_wait_dscnt; no cross-wave sharing.

  v8f g = {};
  for (int c = 0; c < 2; ++c) {
    const __bf16* hp = h + lo * kOPI + c * 32 + hi * 16;
    v8bf h0 = *(const v8bf*)hp;
    v8bf h1 = *(const v8bf*)(hp + 8);
    v16bf ha;
    for (int t = 0; t < 8; ++t) { ha[t] = h0[t]; ha[t + 8] = h1[t]; }
    g = __builtin_amdgcn_wmma_f32_16x16x32_bf16(false, ha, false, w2f[c], (short)0, g, false, false);
  }

  // Store G^T[p][m]: per lane 8 consecutive m values -> one 16B store.
  v8bf gout;
  for (int r = 0; r < 8; ++r) gout[r] = (__bf16)g[r];
  *(v8bf*)(gt + (size_t)lo * kN + m0 + 8 * hi) = gout;
}

// ------ Stage 3: logits tile = A @ G (K-streamed), fused row softmax --------
__global__ __launch_bounds__(256) void gcn_layer2_kernel(
    const float* __restrict__ A, const __bf16* __restrict__ gt,
    float* __restrict__ out) {
  const int wave = threadIdx.x >> 5;
  const int lane = threadIdx.x & 31;
  const int tileIdx = blockIdx.x * kWavesPerBlock + wave;
  if (tileIdx >= kRowTiles) return;                    // wave-uniform exit
  const int m0 = tileIdx * 16;
  const int lo = lane & 15;
  const int hi = lane >> 4;

  v8f acc = {};
  const float* arow = A + (size_t)(m0 + lo) * kN + hi * 16;
  const __bf16* brow = gt + (size_t)lo * kN + hi * 16;
  for (int k0 = 0; k0 < kN; k0 += 32) {
    v16bf af = load_a_frag_f32(arow + k0);
    v16bf bf = *(const v16bf*)(brow + k0);
    acc = __builtin_amdgcn_wmma_f32_16x16x32_bf16(false, af, false, bf, (short)0, acc, false, false);
  }

  // Softmax over the 16 columns: each half-wave's 16 lanes hold one row's cols.
  for (int r = 0; r < 8; ++r) {
    float v = acc[r];
    float mx = v;
    for (int s = 1; s < 16; s <<= 1) mx = fmaxf(mx, __shfl_xor(mx, s, 16));
    float e = __expf(v - mx);
    float sm = e;
    for (int s = 1; s < 16; s <<= 1) sm += __shfl_xor(sm, s, 16);
    out[(size_t)(m0 + r + 8 * hi) * kOPO + lo] = e / sm;
  }
}

extern "C" void kernel_launch(void* const* d_in, const int* in_sizes, int n_in,
                              void* d_out, int out_size, void* d_ws, size_t ws_size,
                              hipStream_t stream) {
  (void)in_sizes; (void)n_in; (void)out_size; (void)ws_size;
  const float* admat    = (const float*)d_in[0];
  const float* features = (const float*)d_in[1];
  const float* W1       = (const float*)d_in[2];
  const float* W2       = (const float*)d_in[3];
  float* out = (float*)d_out;

  __bf16* xw1t = (__bf16*)d_ws;                                   // [64][N] bf16: 2.56 MB
  __bf16* gtb  = (__bf16*)((char*)d_ws + (size_t)kOPI * kN * 2);  // [16][N] bf16: 0.64 MB

  gcn_xw1_kernel<<<kRowTiles, 256, 0, stream>>>(features, W1, xw1t);

  const int blocks = (kRowTiles + kWavesPerBlock - 1) / kWavesPerBlock;  // 157
  gcn_layer1_kernel<<<blocks, 256, 0, stream>>>(admat, xw1t, W2, gtb);
  gcn_layer2_kernel<<<blocks, 256, 0, stream>>>(admat, gtb, out);
}
